// Stage_49873160241240
// MI455X (gfx1250) — matrix operand order repaired and checked by reading, not compiled, not run on hardware
//
#include <hip/hip_runtime.h>

typedef __attribute__((ext_vector_type(16))) _Float16 v16h;
typedef __attribute__((ext_vector_type(8)))  float    v8f;

#define NPTS   100000
#define KNBR   16
#define NKROWS (NPTS * KNBR)
#define EPSBN  1e-5f

// ---- workspace layout (float offsets) ----
#define WS_STATS1  0        // 64  : sum[32], sumsq[32] of h1
#define WS_SC1     64       // 64  : scale[32], shift[32]
#define WS_STATS2  128      // 128 : sum[64], sumsq[64] of h2
#define WS_SC2     256      // 128
#define WS_STATS3  384      // 128 : sum[64], sumsq[64] of hmax
#define WS_SCALE3  512      // 64  : fused BN3*BN4 scale
#define WS_BIAS256 576      // 256 : folded bias @ w_post
#define WS_HMAX    832      // NPTS*64 floats

static __device__ inline float gelu_f(float x) {
  return 0.5f * x * (1.0f + erff(x * 0.70710678118654752f));
}

static __device__ inline v8f wmma16(v16h a, v16h b, v8f c) {
  return __builtin_amdgcn_wmma_f32_16x16x32_f16(false, a, false, b, (short)0, c,
                                                false, false);
}

static __device__ inline v8f zero_acc() {
  v8f z;
#pragma unroll
  for (int i = 0; i < 8; ++i) z[i] = 0.0f;
  return z;
}

// B fragment (16x16x32 f16): lane n = L&15; hi=L>>4; e<8 -> K=hi*8+e ; e>=8 -> K=16+hi*8+(e-8).
// Rows with K >= kreal are zero padding.
static __device__ inline v16h bfrag(const float* __restrict__ w, int ldw, int lane,
                                    int kbase, int ncol, int kreal) {
  const int n = lane & 15, hi = lane >> 4;
  v16h b;
#pragma unroll
  for (int e = 0; e < 16; ++e) {
    const int k = kbase + ((e < 8) ? (hi * 8 + e) : (16 + hi * 8 + (e - 8)));
    b[e] = (k < kreal) ? (_Float16)w[k * ldw + ncol + n] : (_Float16)0.0f;
  }
  return b;
}

// A fragment (16x32 f16) from a row-major LDS stage of 16 x W halves.
// lane L: row m = L&15; hi=L>>4; e<8 -> K=hi*8+e ; e>=8 -> K=16+hi*8+(e-8).
static __device__ inline v16h afrag_lds(const _Float16* st, int W, int lane, int kbase) {
  const int m = lane & 15, hi = lane >> 4;
  const _Float16* row = st + m * W + kbase;
  v16h a;
#pragma unroll
  for (int e = 0; e < 8; ++e) a[e] = row[hi * 8 + e];
#pragma unroll
  for (int e = 0; e < 8; ++e) a[8 + e] = row[16 + hi * 8 + e];
  return a;
}

// Gathered input A fragment: 16 rows (one point's neighbors) x K=32 (7 real feats).
// Only hi==0 lanes carry K=0..7 -> just lanes 0..15 load; rest is zero pad.
static __device__ inline v16h gather_afrag(const float* __restrict__ p,
                                           const float* __restrict__ f,
                                           const int* __restrict__ gidx,
                                           int i, int lane) {
  v16h a;
#pragma unroll
  for (int e = 0; e < 16; ++e) a[e] = (_Float16)0.0f;
  if (lane < 16) {
    const int j = gidx[i * KNBR + lane];
    const float pi0 = p[3 * i], pi1 = p[3 * i + 1], pi2 = p[3 * i + 2];
    a[0] = (_Float16)(p[3 * j]     - pi0);
    a[1] = (_Float16)(p[3 * j + 1] - pi1);
    a[2] = (_Float16)(p[3 * j + 2] - pi2);
    a[3] = (_Float16)f[4 * j];
    a[4] = (_Float16)f[4 * j + 1];
    a[5] = (_Float16)f[4 * j + 2];
    a[6] = (_Float16)f[4 * j + 3];
  }
  return a;
}

// Accumulate per-channel sum/sumsq from a 16x16 f32 D fragment into LDS.
static __device__ inline void stats_accum(v8f d, float* acc, int lane, int noff, int C) {
  float sm = 0.f, sq = 0.f;
#pragma unroll
  for (int r = 0; r < 8; ++r) { const float v = d[r]; sm += v; sq += v * v; }
  sm += __shfl_xor(sm, 16, 32);
  sq += __shfl_xor(sq, 16, 32);
  if (lane < 16) {
    atomicAdd(&acc[noff + lane], sm);
    atomicAdd(&acc[C + noff + lane], sq);
  }
}

// Apply BN (scale s, shift t) + exact GELU to D fragment, store f16 to LDS stage.
static __device__ inline void bn_gelu_store(v8f d, _Float16* st, int W, int lane,
                                            int nbase, float s, float t) {
  const int n = lane & 15, hi = lane >> 4;
#pragma unroll
  for (int r = 0; r < 8; ++r) {
    const float y = gelu_f(d[r] * s + t);
    st[(hi * 8 + r) * W + nbase + n] = (_Float16)y;
  }
}

// ---------------- kernels ----------------

__global__ void k_zero(float* __restrict__ ws) {
  const int t = threadIdx.x;
  if (t < 64)  ws[WS_STATS1 + t] = 0.f;
  if (t < 128) ws[WS_STATS2 + t] = 0.f;
  if (t < 128) ws[WS_STATS3 + t] = 0.f;
}

// Pass 1: gather + Linear1 (WMMA), accumulate BN1 batch stats. No stores of h1.
__global__ __launch_bounds__(256) void k_stage1(const float* __restrict__ p,
                                                const float* __restrict__ f,
                                                const int* __restrict__ gidx,
                                                const float* __restrict__ w1,
                                                float* __restrict__ ws) {
  __shared__ float acc[64];
  const int lane = threadIdx.x & 31;
  const int wave = threadIdx.x >> 5;
  for (int t = threadIdx.x; t < 64; t += 256) acc[t] = 0.f;
  __syncthreads();

  v16h b1f0 = bfrag(w1, 32, lane, 0, 0, 7);
  v16h b1f1 = bfrag(w1, 32, lane, 0, 16, 7);

  const int gw = blockIdx.x * 8 + wave, nw = gridDim.x * 8;
  for (int i = gw; i < NPTS; i += nw) {
    v16h a1 = gather_afrag(p, f, gidx, i, lane);
    v8f d0 = wmma16(a1, b1f0, zero_acc());
    v8f d1 = wmma16(a1, b1f1, zero_acc());
    stats_accum(d0, acc, lane, 0, 32);
    stats_accum(d1, acc, lane, 16, 32);
  }
  __syncthreads();
  for (int t = threadIdx.x; t < 64; t += 256) atomicAdd(ws + WS_STATS1 + t, acc[t]);
}

// Finalize BN scale/shift from sum/sumsq (biased var, matches jnp.var).
__global__ void k_fin12(const float* __restrict__ st, const float* __restrict__ g,
                        const float* __restrict__ b, float* __restrict__ outsc,
                        int C, float cnt) {
  const int c = threadIdx.x;
  if (c < C) {
    const float mean = st[c] / cnt;
    const float var  = st[C + c] / cnt - mean * mean;
    const float sc   = g[c] * rsqrtf(var + EPSBN);
    outsc[c]     = sc;
    outsc[C + c] = b[c] - mean * sc;
  }
}

// Pass 2: recompute L1, BN1+GELU, Linear2 (WMMA), accumulate BN2 stats.
__global__ __launch_bounds__(256) void k_stage2(const float* __restrict__ p,
                                                const float* __restrict__ f,
                                                const int* __restrict__ gidx,
                                                const float* __restrict__ w1,
                                                const float* __restrict__ w2,
                                                float* __restrict__ ws) {
  __shared__ float acc[128];
  __shared__ _Float16 stage1[8][16][32];
  const int lane = threadIdx.x & 31;
  const int wave = threadIdx.x >> 5;
  for (int t = threadIdx.x; t < 128; t += 256) acc[t] = 0.f;
  __syncthreads();

  const float* sc1 = ws + WS_SC1;
  const int nl = lane & 15;
  const float s1a = sc1[nl],      t1a = sc1[32 + nl];
  const float s1b = sc1[16 + nl], t1b = sc1[48 + nl];

  v16h b1f0 = bfrag(w1, 32, lane, 0, 0, 7);
  v16h b1f1 = bfrag(w1, 32, lane, 0, 16, 7);
  v16h b2f[4];
#pragma unroll
  for (int nt = 0; nt < 4; ++nt) b2f[nt] = bfrag(w2, 64, lane, 0, nt * 16, 32);

  _Float16* st = &stage1[wave][0][0];
  const int gw = blockIdx.x * 8 + wave, nw = gridDim.x * 8;
  for (int i = gw; i < NPTS; i += nw) {
    v16h a1 = gather_afrag(p, f, gidx, i, lane);
    v8f d0 = wmma16(a1, b1f0, zero_acc());
    v8f d1 = wmma16(a1, b1f1, zero_acc());
    bn_gelu_store(d0, st, 32, lane, 0, s1a, t1a);
    bn_gelu_store(d1, st, 32, lane, 16, s1b, t1b);
    v16h a2 = afrag_lds(st, 32, lane, 0);
#pragma unroll
    for (int nt = 0; nt < 4; ++nt) {
      v8f d2 = wmma16(a2, b2f[nt], zero_acc());
      stats_accum(d2, acc, lane, nt * 16, 64);
    }
  }
  __syncthreads();
  for (int t = threadIdx.x; t < 128; t += 256) atomicAdd(ws + WS_STATS2 + t, acc[t]);
}

// Pass 3: full chain through Linear3, max-pool over K inside D layout, store hmax,
// accumulate stats for the fused trailing BNs.
__global__ __launch_bounds__(256) void k_stage3(const float* __restrict__ p,
                                                const float* __restrict__ f,
                                                const int* __restrict__ gidx,
                                                const float* __restrict__ w1,
                                                const float* __restrict__ w2,
                                                const float* __restrict__ w3,
                                                float* __restrict__ ws) {
  __shared__ float acc[128];
  __shared__ _Float16 stage1[8][16][32];
  __shared__ _Float16 stage2[8][16][64];
  const int lane = threadIdx.x & 31;
  const int wave = threadIdx.x >> 5;
  for (int t = threadIdx.x; t < 128; t += 256) acc[t] = 0.f;
  __syncthreads();

  const float* sc1 = ws + WS_SC1;
  const float* sc2 = ws + WS_SC2;
  const int nl = lane & 15;
  const float s1a = sc1[nl],      t1a = sc1[32 + nl];
  const float s1b = sc1[16 + nl], t1b = sc1[48 + nl];
  float s2[4], t2[4];
#pragma unroll
  for (int nt = 0; nt < 4; ++nt) {
    s2[nt] = sc2[nt * 16 + nl];
    t2[nt] = sc2[64 + nt * 16 + nl];
  }

  v16h b1f0 = bfrag(w1, 32, lane, 0, 0, 7);
  v16h b1f1 = bfrag(w1, 32, lane, 0, 16, 7);
  v16h b2f[4];
#pragma unroll
  for (int nt = 0; nt < 4; ++nt) b2f[nt] = bfrag(w2, 64, lane, 0, nt * 16, 32);
  v16h b3f[2][4];
#pragma unroll
  for (int ks = 0; ks < 2; ++ks)
#pragma unroll
    for (int nt = 0; nt < 4; ++nt)
      b3f[ks][nt] = bfrag(w3, 64, lane, ks * 32, nt * 16, 64);

  _Float16* st1 = &stage1[wave][0][0];
  _Float16* st2 = &stage2[wave][0][0];
  float* hmax = ws + WS_HMAX;
  const int gw = blockIdx.x * 8 + wave, nw = gridDim.x * 8;
  for (int i = gw; i < NPTS; i += nw) {
    v16h a1 = gather_afrag(p, f, gidx, i, lane);
    v8f d0 = wmma16(a1, b1f0, zero_acc());
    v8f d1 = wmma16(a1, b1f1, zero_acc());
    bn_gelu_store(d0, st1, 32, lane, 0, s1a, t1a);
    bn_gelu_store(d1, st1, 32, lane, 16, s1b, t1b);
    v16h a2 = afrag_lds(st1, 32, lane, 0);
#pragma unroll
    for (int nt = 0; nt < 4; ++nt) {
      v8f d2 = wmma16(a2, b2f[nt], zero_acc());
      bn_gelu_store(d2, st2, 64, lane, nt * 16, s2[nt], t2[nt]);
    }
    v16h a3lo = afrag_lds(st2, 64, lane, 0);
    v16h a3hi = afrag_lds(st2, 64, lane, 32);
#pragma unroll
    for (int nt = 0; nt < 4; ++nt) {
      v8f d3 = wmma16(a3lo, b3f[0][nt], zero_acc());
      d3 = wmma16(a3hi, b3f[1][nt], d3);
      float mx = d3[0];
#pragma unroll
      for (int r = 1; r < 8; ++r) mx = fmaxf(mx, d3[r]);
      mx = fmaxf(mx, __shfl_xor(mx, 16, 32));  // combine both row halves
      if (lane < 16) {
        hmax[i * 64 + nt * 16 + lane] = mx;
        atomicAdd(&acc[nt * 16 + lane], mx);
        atomicAdd(&acc[64 + nt * 16 + lane], mx * mx);
      }
    }
  }
  __syncthreads();
  for (int t = threadIdx.x; t < 128; t += 256) atomicAdd(ws + WS_STATS3 + t, acc[t]);
}

// Fuse nbr_bn + post BN into per-channel affine (nbr_b cancels exactly since the
// mean of BN1's output is nbr_b), then fold the bias through w_post -> bias256.
__global__ void k_fin3(const float* __restrict__ st, const float* __restrict__ nbr_g,
                       const float* __restrict__ nbr_b, const float* __restrict__ post_g,
                       const float* __restrict__ post_b, const float* __restrict__ w_post,
                       float* __restrict__ scale3, float* __restrict__ bias256) {
  (void)nbr_b;
  __shared__ float biasc[64];
  const int t = threadIdx.x;
  if (t < 64) {
    const float mean = st[t] / (float)NPTS;
    const float var  = st[64 + t] / (float)NPTS - mean * mean;
    const float a1c  = nbr_g[t] * rsqrtf(var + EPSBN);
    const float sc   = a1c * post_g[t] * rsqrtf(a1c * a1c * var + EPSBN);
    scale3[t] = sc;
    biasc[t]  = post_b[t] - mean * sc;
  }
  __syncthreads();
  float s = 0.f;
  for (int c = 0; c < 64; ++c) s += biasc[c] * w_post[c * 256 + t];
  bias256[t] = s;
}

// Final GEMM: out = (hmax * scale3) @ w_post + bias256.  Bias loaded into C.
__global__ __launch_bounds__(256) void k_out(const float* __restrict__ hmax,
                                             const float* __restrict__ w_post,
                                             const float* __restrict__ scale3,
                                             const float* __restrict__ bias256,
                                             float* __restrict__ out) {
  __shared__ float s_sc[64];
  __shared__ float s_bias[256];
  const int lane = threadIdx.x & 31, wave = threadIdx.x >> 5;
  if (threadIdx.x < 64) s_sc[threadIdx.x] = scale3[threadIdx.x];
  s_bias[threadIdx.x] = bias256[threadIdx.x];
  __syncthreads();

  const int tile = blockIdx.x * 8 + wave;
  if (tile >= NPTS / 16) return;
  const int row0 = tile * 16;
  const int m = lane & 15, hi = lane >> 4;
  const float* hrow = hmax + (row0 + m) * 64;

  v16h a0, a1;
#pragma unroll
  for (int e = 0; e < 8; ++e) { const int k = hi * 8 + e;      a0[e]     = (_Float16)(hrow[k] * s_sc[k]); }
#pragma unroll
  for (int e = 0; e < 8; ++e) { const int k = 16 + hi * 8 + e; a0[8 + e] = (_Float16)(hrow[k] * s_sc[k]); }
#pragma unroll
  for (int e = 0; e < 8; ++e) { const int k = 32 + hi * 8 + e; a1[e]     = (_Float16)(hrow[k] * s_sc[k]); }
#pragma unroll
  for (int e = 0; e < 8; ++e) { const int k = 48 + hi * 8 + e; a1[8 + e] = (_Float16)(hrow[k] * s_sc[k]); }

#pragma unroll 4
  for (int nt = 0; nt < 16; ++nt) {
    v16h b0 = bfrag(w_post, 256, lane, 0, nt * 16, 64);
    v16h b1 = bfrag(w_post, 256, lane, 32, nt * 16, 64);
    const float bb = s_bias[nt * 16 + m];
    v8f c;
#pragma unroll
    for (int r = 0; r < 8; ++r) c[r] = bb;
    c = wmma16(a0, b0, c);
    c = wmma16(a1, b1, c);
#pragma unroll
    for (int r = 0; r < 8; ++r)
      out[(row0 + hi * 8 + r) * 256 + nt * 16 + m] = c[r];
  }
}

extern "C" void kernel_launch(void* const* d_in, const int* in_sizes, int n_in,
                              void* d_out, int out_size, void* d_ws, size_t ws_size,
                              hipStream_t stream) {
  (void)in_sizes; (void)n_in; (void)out_size; (void)ws_size;
  const float* p      = (const float*)d_in[0];
  const float* f      = (const float*)d_in[1];
  const int*   gidx   = (const int*)d_in[2];
  const float* w1     = (const float*)d_in[3];
  const float* g1     = (const float*)d_in[4];
  const float* b1     = (const float*)d_in[5];
  const float* w2     = (const float*)d_in[6];
  const float* g2     = (const float*)d_in[7];
  const float* b2     = (const float*)d_in[8];
  const float* w3     = (const float*)d_in[9];
  const float* nbr_g  = (const float*)d_in[10];
  const float* nbr_b  = (const float*)d_in[11];
  const float* post_g = (const float*)d_in[12];
  const float* post_b = (const float*)d_in[13];
  const float* w_post = (const float*)d_in[14];
  float* out = (float*)d_out;
  float* ws  = (float*)d_ws;

  k_zero<<<1, 512, 0, stream>>>(ws);
  k_stage1<<<1024, 256, 0, stream>>>(p, f, gidx, w1, ws);
  k_fin12<<<1, 64, 0, stream>>>(ws + WS_STATS1, g1, b1, ws + WS_SC1, 32, (float)NKROWS);
  k_stage2<<<1024, 256, 0, stream>>>(p, f, gidx, w1, w2, ws);
  k_fin12<<<1, 64, 0, stream>>>(ws + WS_STATS2, g2, b2, ws + WS_SC2, 64, (float)NKROWS);
  k_stage3<<<1024, 256, 0, stream>>>(p, f, gidx, w1, w2, w3, ws);
  k_fin3<<<1, 256, 0, stream>>>(ws + WS_STATS3, nbr_g, nbr_b, post_g, post_b, w_post,
                                ws + WS_SCALE3, ws + WS_BIAS256);
  k_out<<<(NPTS / 16 + 7) / 8, 256, 0, stream>>>(ws + WS_HMAX, w_post, ws + WS_SCALE3,
                                                 ws + WS_BIAS256, out);
}
